// MoELayer_8289286881673
// MI455X (gfx1250) — matrix operand order repaired and checked by reading, not compiled, not run on hardware
//
#include <hip/hip_runtime.h>
#include <hip/hip_bf16.h>
#include <stdint.h>

#define T_TOK 8192
#define DM    1024
#define DF    4096
#define NE    8
#define ROWS_MAX (T_TOK * 2)

// LDS tile row stride: 40 bf16 = 80B. 80B stride => bank = 20*row mod 64,
// 16 distinct banks across a 16-lane fragment read group (conflict-free),
// while every 16B chunk stays 16B-aligned for async stores / ds_load_b128.
#define LROW   40
#define LROWB  (LROW * 2)
#define BUFB   (128 * LROWB)   // 10240 B per tile buffer

typedef __attribute__((ext_vector_type(16))) __bf16 v16bf;
typedef __attribute__((ext_vector_type(8)))  float  v8f;

struct Frag { uint4 u[2]; };

static __device__ __forceinline__ v16bf frag_cast(const Frag& f) {
  union U { Frag f; v16bf v; } u;
  u.f = f;
  return u.v;
}

// f32 -> bf16, round-to-nearest-even
static __device__ __forceinline__ uint16_t f2bf(float x) {
  uint32_t u = __float_as_uint(x);
  u += 0x7fffu + ((u >> 16) & 1u);
  return (uint16_t)(u >> 16);
}

// Issue two async 16B global->LDS copies (one 32B half-row). Tracked by
// ASYNCcnt; LDS byte address comes from truncating the flat shared pointer.
static __device__ __forceinline__ void async_b128_pair(const void* src,
                                                       uint32_t ldsByte)
{
  asm volatile(
      "global_load_async_to_lds_b128 %0, %1, off\n\t"
      "global_load_async_to_lds_b128 %2, %1, off offset:16"
      :
      : "v"(ldsByte), "v"(src), "v"(ldsByte + 16)
      : "memory");
}

// ---------------------------------------------------------------------------
// Small setup kernels
// ---------------------------------------------------------------------------
__global__ void init_small(int* counts, int* fill, float* usage) {
  int i = threadIdx.x;
  if (i < NE) { counts[i] = 0; fill[i] = 0; usage[i] = 0.f; }
}

__global__ void zero_out(float* __restrict__ out) {
  size_t i = (size_t)blockIdx.x * blockDim.x + threadIdx.x;
  out[i] = 0.f;
}

// ---------------------------------------------------------------------------
// One-time weight conversion: W f32 [e][KD][NF] -> Wt bf16 [e][NF][KD]
// (transposed so GEMM B tiles are contiguous 64B rows along K -> async-copyable)
// ---------------------------------------------------------------------------
template<int KD, int NF>
__global__ void __launch_bounds__(256) convert_w_kernel(
    const float* __restrict__ W, uint16_t* __restrict__ Wt)
{
  __shared__ alignas(16) uint16_t lt[128 * 32];
  const int e   = blockIdx.z;
  const int k0  = blockIdx.y * 32;
  const int n0  = blockIdx.x * 128;
  const int tid = threadIdx.x;
  const float* We = W + (size_t)e * KD * NF;

#pragma unroll
  for (int it = 0; it < 4; ++it) {
    int idx = tid + it * 256;        // 0..1023
    int kk  = idx >> 5;              // 0..31
    int nn  = (idx & 31) * 4;        // 0..124
    float4 w = *(const float4*)(We + (size_t)(k0 + kk) * NF + n0 + nn);
    lt[(nn + 0) * 32 + kk] = f2bf(w.x);
    lt[(nn + 1) * 32 + kk] = f2bf(w.y);
    lt[(nn + 2) * 32 + kk] = f2bf(w.z);
    lt[(nn + 3) * 32 + kk] = f2bf(w.w);
  }
  __syncthreads();

  const int n  = tid >> 1;           // 0..127
  const int hs = tid & 1;            // 16-element half of the k-row
  uint16_t* dst = Wt + ((size_t)e * NF + n0 + n) * KD + k0 + hs * 16;
  const uint4* srcp = (const uint4*)lt + n * 4 + hs * 2;
  ((uint4*)dst)[0] = srcp[0];
  ((uint4*)dst)[1] = srcp[1];
}

// ---------------------------------------------------------------------------
// Gating: one wave32 per token. 8 dot products of length 1024, shfl reduce,
// full softmax -> usage (aux loss), top-2 softmax -> routing weights.
// ---------------------------------------------------------------------------
__global__ void __launch_bounds__(256) gate_kernel(
    const float* __restrict__ x, const float* __restrict__ Wg,
    const float* __restrict__ bg, int* __restrict__ counts,
    float* __restrict__ usage, int* __restrict__ tok_e,
    float2* __restrict__ tok_w)
{
  __shared__ float s_usage[NE];
  if (threadIdx.x < NE) s_usage[threadIdx.x] = 0.f;
  __syncthreads();

  const int wave = threadIdx.x >> 5;
  const int lane = threadIdx.x & 31;
  const int t = blockIdx.x * 8 + wave;

  float acc[NE];
#pragma unroll
  for (int e = 0; e < NE; ++e) acc[e] = 0.f;

  const float* xr = x + (size_t)t * DM;
  for (int d = lane; d < DM; d += 32) {
    float xv = xr[d];
    const float4* wp = (const float4*)(Wg + (size_t)d * NE);
    float4 w0 = wp[0], w1 = wp[1];
    acc[0] += xv * w0.x; acc[1] += xv * w0.y;
    acc[2] += xv * w0.z; acc[3] += xv * w0.w;
    acc[4] += xv * w1.x; acc[5] += xv * w1.y;
    acc[6] += xv * w1.z; acc[7] += xv * w1.w;
  }
#pragma unroll
  for (int e = 0; e < NE; ++e) {
#pragma unroll
    for (int off = 16; off > 0; off >>= 1)
      acc[e] += __shfl_xor(acc[e], off, 32);
  }

  if (lane == 0) {
    float lg[NE];
#pragma unroll
    for (int e = 0; e < NE; ++e) lg[e] = acc[e] + bg[e];

    float mx = lg[0];
#pragma unroll
    for (int e = 1; e < NE; ++e) mx = fmaxf(mx, lg[e]);
    float ex[NE];
    float se = 0.f;
#pragma unroll
    for (int e = 0; e < NE; ++e) { ex[e] = __expf(lg[e] - mx); se += ex[e]; }
    float inv = 1.f / se;
#pragma unroll
    for (int e = 0; e < NE; ++e) atomicAdd(&s_usage[e], ex[e] * inv);

    int i1 = 0;
#pragma unroll
    for (int e = 1; e < NE; ++e) if (lg[e] > lg[i1]) i1 = e;
    int i2 = (i1 == 0) ? 1 : 0;
#pragma unroll
    for (int e = 0; e < NE; ++e) if (e != i1 && lg[e] > lg[i2]) i2 = e;

    float m2  = fmaxf(lg[i1], lg[i2]);
    float e1v = __expf(lg[i1] - m2);
    float e2v = __expf(lg[i2] - m2);
    float sw  = e1v + e2v;

    tok_e[t] = i1 | (i2 << 8);
    tok_w[t] = make_float2(e1v / sw, e2v / sw);
    atomicAdd(&counts[i1], 1);
    atomicAdd(&counts[i2], 1);
  }
  __syncthreads();
  if (threadIdx.x < NE) atomicAdd(&usage[threadIdx.x], s_usage[threadIdx.x]);
}

// ---------------------------------------------------------------------------
// Prefix-sum of expert counts + aux loss
// ---------------------------------------------------------------------------
__global__ void scan_kernel(const int* __restrict__ counts, int* __restrict__ offsets,
                            const float* __restrict__ usage, float* __restrict__ aux)
{
  if (threadIdx.x == 0) {
    int s = 0;
    for (int e = 0; e < NE; ++e) { offsets[e] = s; s += counts[e]; }
    offsets[NE] = s;
    float a = 0.f;
    for (int e = 0; e < NE; ++e) {
      float u = usage[e] / (float)T_TOK;
      a += u * u;
    }
    *aux = a * (float)NE;
  }
}

// ---------------------------------------------------------------------------
// Compact routed rows into bf16 X-pool. One wave per (token, k) pair.
// ---------------------------------------------------------------------------
__global__ void __launch_bounds__(256) gather_kernel(
    const float* __restrict__ x, const int* __restrict__ tok_e,
    const float2* __restrict__ tok_w, const int* __restrict__ offsets,
    int* __restrict__ fill, int* __restrict__ row_token,
    float* __restrict__ row_w, uint16_t* __restrict__ Xpool)
{
  __shared__ int s_pos[8];
  const int wave = threadIdx.x >> 5;
  const int lane = threadIdx.x & 31;
  const int wid = blockIdx.x * 8 + wave;   // 0 .. ROWS_MAX-1
  const int t = wid >> 1;
  const int k = wid & 1;

  if (lane == 0) {
    int ee = tok_e[t];
    int e = k ? ((ee >> 8) & 0xff) : (ee & 0xff);
    float2 w2 = tok_w[t];
    int pos = offsets[e] + atomicAdd(&fill[e], 1);
    row_token[pos] = t;
    row_w[pos] = k ? w2.y : w2.x;
    s_pos[wave] = pos;
  }
  __syncthreads();
  const int pos = s_pos[wave];

  const float4* src = (const float4*)(x + (size_t)t * DM);
  uint64_t* dst = (uint64_t*)(Xpool + (size_t)pos * DM);
  for (int i = lane; i < DM / 4; i += 32) {
    float4 v = src[i];
    uint64_t pk = (uint64_t)f2bf(v.x)
                | ((uint64_t)f2bf(v.y) << 16)
                | ((uint64_t)f2bf(v.z) << 32)
                | ((uint64_t)f2bf(v.w) << 48);
    dst[i] = pk;
  }
}

// ---------------------------------------------------------------------------
// WMMA GEMM mainloop: C[128x128] tile, 8 waves (4M x 2N), K-step 32, bf16.
// A: bf16 pool rows [rows][KDIM]. B: bf16 transposed weights [NFULL][KDIM].
// Both tiles staged [row][k] with 80B padded row stride in LDS via
// double-buffered GLOBAL_LOAD_ASYNC_TO_LDS_B128 (ASYNCcnt, in-order
// completion: after issuing tile k+1's 4 instructions, s_wait_asynccnt 4 =>
// tile k resident). A rows are clamped (never predicated) so EXEC stays
// all-ones for WMMA; fragment ds_load_b128s are bank-conflict-free.
// ---------------------------------------------------------------------------
template<int KDIM>
static __device__ __forceinline__ void moe_gemm_tile(
    const uint16_t* __restrict__ Apool, const uint16_t* __restrict__ Bmat,
    int r0, int rows, int mBase, int nBase,
    uint16_t* lA, uint16_t* lB, v8f acc[2][4])
{
  const int tid   = threadIdx.x;
  const int lane  = tid & 31;
  const int wave  = tid >> 5;
  const int waveM = wave >> 1;   // 0..3
  const int waveN = wave & 1;    // 0..1
  const int half  = lane >> 4;   // K-half select
  const int ml    = lane & 15;

  // per-thread staging assignment: 2 x 16B (one 32B half-row) per tile each
  const int r    = tid >> 1;               // 0..127
  const int hs   = tid & 1;                // 16-element half
  const int rmax = rows - mBase - 1;       // >= 0 (block early-exits otherwise)
  const int ra   = r < rmax ? r : rmax;    // clamp OOB rows (results discarded)
  const uint16_t* aRow = Apool + (size_t)(r0 + mBase + ra) * KDIM + hs * 16;
  const uint16_t* bRow = Bmat + (size_t)(nBase + r) * KDIM + hs * 16;
  const uint32_t ldsOff = (uint32_t)(r * LROWB + hs * 32);
  const uint32_t lAbase = (uint32_t)(uintptr_t)lA;
  const uint32_t lBbase = (uint32_t)(uintptr_t)lB;

  constexpr int KSTEPS = KDIM / 32;

  // prologue: stage tile 0 into buffer 0
  async_b128_pair(aRow, lAbase + ldsOff);
  async_b128_pair(bRow, lBbase + ldsOff);

  for (int ks = 0; ks < KSTEPS; ++ks) {
    const int cur = ks & 1;
    if (ks + 1 < KSTEPS) {
      const uint32_t nb = (uint32_t)(cur ^ 1) * BUFB;
      async_b128_pair(aRow + (ks + 1) * 32, lAbase + nb + ldsOff);
      async_b128_pair(bRow + (ks + 1) * 32, lBbase + nb + ldsOff);
      asm volatile("s_wait_asynccnt 0x4" ::: "memory");
    } else {
      asm volatile("s_wait_asynccnt 0x0" ::: "memory");
    }
    __syncthreads();

    const uint16_t* bufA = lA + cur * (BUFB / 2);
    const uint16_t* bufB = lB + cur * (BUFB / 2);

    Frag af[2];
#pragma unroll
    for (int i = 0; i < 2; ++i) {
      int m = waveM * 32 + i * 16 + ml;
      const uint4* ap = (const uint4*)bufA;
      af[i].u[0] = ap[m * 5 + half];        // K = half*8 .. +7
      af[i].u[1] = ap[m * 5 + 2 + half];    // K = 16 + half*8 .. +7
    }
    Frag bfr[4];
#pragma unroll
    for (int j = 0; j < 4; ++j) {
      int n = waveN * 64 + j * 16 + ml;
      const uint4* bp = (const uint4*)bufB;
      bfr[j].u[0] = bp[n * 5 + half];          // K = half*8 .. +7
      bfr[j].u[1] = bp[n * 5 + 2 + half];      // K = 16 + half*8 .. +7
    }
#pragma unroll
    for (int i = 0; i < 2; ++i) {
      v16bf av = frag_cast(af[i]);
#pragma unroll
      for (int j = 0; j < 4; ++j) {
        acc[i][j] = __builtin_amdgcn_wmma_f32_16x16x32_bf16(
            false, av, false, frag_cast(bfr[j]), (short)0, acc[i][j],
            false, false);
      }
    }
    __syncthreads();   // all waves done reading before this buffer is refilled
  }
}

// ---------------------------------------------------------------------------
// FFN layer 1: H = relu(Xe @ W1[e] + b1[e]), stored bf16
// ---------------------------------------------------------------------------
__global__ void __launch_bounds__(256) ffn1_kernel(
    const uint16_t* __restrict__ Xpool, const uint16_t* __restrict__ W1t,
    const float* __restrict__ b1, const int* __restrict__ offsets,
    uint16_t* __restrict__ Hpool)
{
  __shared__ alignas(16) uint16_t lA[2 * 128 * LROW];
  __shared__ alignas(16) uint16_t lB[2 * 128 * LROW];

  const int e    = blockIdx.z;
  const int r0   = offsets[e];
  const int rows = offsets[e + 1] - r0;
  const int mBase = blockIdx.y * 128;
  if (mBase >= rows) return;   // uniform per-block: EXEC stays full for WMMA
  const int nBase = blockIdx.x * 128;

  v8f acc[2][4] = {};
  moe_gemm_tile<DM>(Xpool, W1t + (size_t)e * DF * DM, r0, rows, mBase, nBase,
                    lA, lB, acc);

  const int lane  = threadIdx.x & 31;
  const int wave  = threadIdx.x >> 5;
  const int waveM = wave >> 1, waveN = wave & 1;
  const int half  = lane >> 4, nl = lane & 15;
  const float* b1e = b1 + (size_t)e * DF;

  float bv[4];
#pragma unroll
  for (int j = 0; j < 4; ++j) bv[j] = b1e[nBase + waveN * 64 + j * 16 + nl];

#pragma unroll
  for (int i = 0; i < 2; ++i) {
#pragma unroll
    for (int r = 0; r < 8; ++r) {
      int mlcl = mBase + waveM * 32 + i * 16 + r + 8 * half;
      if (mlcl < rows) {
        uint16_t* hrow = Hpool + (size_t)(r0 + mlcl) * DF;
#pragma unroll
        for (int j = 0; j < 4; ++j) {
          int n = nBase + waveN * 64 + j * 16 + nl;
          float h = acc[i][j][r] + bv[j];
          h = h > 0.f ? h : 0.f;
          hrow[n] = f2bf(h);
        }
      }
    }
  }
}

// ---------------------------------------------------------------------------
// FFN layer 2: out[token] += w * (H @ W2[e] + b2[e])
// ---------------------------------------------------------------------------
__global__ void __launch_bounds__(256) ffn2_kernel(
    const uint16_t* __restrict__ Hpool, const uint16_t* __restrict__ W2t,
    const float* __restrict__ b2, const int* __restrict__ offsets,
    const int* __restrict__ row_token, const float* __restrict__ row_w,
    float* __restrict__ out)
{
  __shared__ alignas(16) uint16_t lA[2 * 128 * LROW];
  __shared__ alignas(16) uint16_t lB[2 * 128 * LROW];

  const int e    = blockIdx.z;
  const int r0   = offsets[e];
  const int rows = offsets[e + 1] - r0;
  const int mBase = blockIdx.y * 128;
  if (mBase >= rows) return;
  const int nBase = blockIdx.x * 128;

  v8f acc[2][4] = {};
  moe_gemm_tile<DF>(Hpool, W2t + (size_t)e * DM * DF, r0, rows, mBase, nBase,
                    lA, lB, acc);

  const int lane  = threadIdx.x & 31;
  const int wave  = threadIdx.x >> 5;
  const int waveM = wave >> 1, waveN = wave & 1;
  const int half  = lane >> 4, nl = lane & 15;
  const float* b2e = b2 + (size_t)e * DM;

  float bv[4];
#pragma unroll
  for (int j = 0; j < 4; ++j) bv[j] = b2e[nBase + waveN * 64 + j * 16 + nl];

#pragma unroll
  for (int i = 0; i < 2; ++i) {
#pragma unroll
    for (int r = 0; r < 8; ++r) {
      int mlcl = mBase + waveM * 32 + i * 16 + r + 8 * half;
      if (mlcl < rows) {
        int row = r0 + mlcl;
        float w = row_w[row];
        int tok = row_token[row];
        float* orow = out + (size_t)tok * DM;
#pragma unroll
        for (int j = 0; j < 4; ++j) {
          int n = nBase + waveN * 64 + j * 16 + nl;
          atomicAdd(&orow[n], w * (acc[i][j][r] + bv[j]));
        }
      }
    }
  }
}

// ---------------------------------------------------------------------------
// Host launch
// ---------------------------------------------------------------------------
extern "C" void kernel_launch(void* const* d_in, const int* in_sizes, int n_in,
                              void* d_out, int out_size, void* d_ws, size_t ws_size,
                              hipStream_t stream)
{
  (void)in_sizes; (void)n_in; (void)out_size; (void)ws_size;

  const float* x  = (const float*)d_in[0];
  const float* Wg = (const float*)d_in[1];
  const float* bg = (const float*)d_in[2];
  const float* W1 = (const float*)d_in[3];
  const float* b1 = (const float*)d_in[4];
  const float* W2 = (const float*)d_in[5];
  const float* b2 = (const float*)d_in[6];

  float* out = (float*)d_out;
  float* aux = out + (size_t)T_TOK * DM;

  uint8_t* p = (uint8_t*)d_ws;
  auto carve = [&p](size_t bytes) -> uint8_t* {
    uint8_t* q = p;
    p += (bytes + 255) & ~(size_t)255;
    return q;
  };
  int*      counts    = (int*)carve(NE * sizeof(int));
  int*      fill      = (int*)carve(NE * sizeof(int));
  int*      offsets   = (int*)carve((NE + 1) * sizeof(int));
  float*    usage     = (float*)carve(NE * sizeof(float));
  int*      tok_e     = (int*)carve((size_t)T_TOK * sizeof(int));
  float2*   tok_w     = (float2*)carve((size_t)T_TOK * sizeof(float2));
  int*      row_token = (int*)carve((size_t)ROWS_MAX * sizeof(int));
  float*    row_w     = (float*)carve((size_t)ROWS_MAX * sizeof(float));
  uint16_t* Xpool     = (uint16_t*)carve((size_t)ROWS_MAX * DM * 2);
  uint16_t* Hpool     = (uint16_t*)carve((size_t)ROWS_MAX * DF * 2);
  uint16_t* W1t       = (uint16_t*)carve((size_t)NE * DM * DF * 2);
  uint16_t* W2t       = (uint16_t*)carve((size_t)NE * DF * DM * 2);

  init_small<<<1, 64, 0, stream>>>(counts, fill, usage);
  zero_out<<<(T_TOK * DM) / 256, 256, 0, stream>>>(out);

  // one-time weight convert + transpose to bf16 [e][n][k]
  convert_w_kernel<DM, DF><<<dim3(DF / 128, DM / 32, NE), 256, 0, stream>>>(W1, W1t);
  convert_w_kernel<DF, DM><<<dim3(DM / 128, DF / 32, NE), 256, 0, stream>>>(W2, W2t);

  gate_kernel<<<T_TOK / 8, 256, 0, stream>>>(x, Wg, bg, counts, usage, tok_e, tok_w);
  scan_kernel<<<1, 32, 0, stream>>>(counts, offsets, usage, aux);
  gather_kernel<<<ROWS_MAX / 8, 256, 0, stream>>>(x, tok_e, tok_w, offsets, fill,
                                                  row_token, row_w, Xpool);
  ffn1_kernel<<<dim3(DF / 128, ROWS_MAX / 128, NE), 256, 0, stream>>>(
      Xpool, W1t, b1, offsets, Hpool);
  ffn2_kernel<<<dim3(DM / 128, ROWS_MAX / 128, NE), 256, 0, stream>>>(
      Hpool, W2t, b2, offsets, row_token, row_w, out);
}
